// NodeMLP_11991548690715
// MI455X (gfx1250) — matrix operand order repaired and checked by reading, not compiled, not run on hardware
//
#include <hip/hip_runtime.h>
#include <hip/hip_bf16.h>
#include <stdint.h>

#define NN  40962
#define NE  327680
#define D   512
#define IND 1024

typedef __bf16 v16bf __attribute__((ext_vector_type(16)));
typedef float  v8f   __attribute__((ext_vector_type(8)));
typedef float  v4f   __attribute__((ext_vector_type(4)));
typedef unsigned int v4u __attribute__((ext_vector_type(4)));

union AF { v16bf v; v4u q[2]; };
union BF16CV { unsigned short s; __bf16 b; };

__device__ __forceinline__ unsigned short f2bfbits(float f) {
    union { float f; unsigned u; } c; c.f = f;
    unsigned r = c.u + 0x7FFFu + ((c.u >> 16) & 1u);   // round-to-nearest-even
    return (unsigned short)(r >> 16);
}

// ---------------- kernel 1: zero the edge_sum accumulator ----------------
__global__ __launch_bounds__(256) void zero_f32(float* __restrict__ p) {
    size_t i = ((size_t)blockIdx.x * 256 + threadIdx.x) * 4;
    if (i < (size_t)NN * D) {
        v4f z; z[0] = 0.f; z[1] = 0.f; z[2] = 0.f; z[3] = 0.f;
        *(v4f*)(p + i) = z;
    }
}

// ---------------- kernel 2: scatter-add edges (hw f32 atomics) -----------
__global__ __launch_bounds__(128) void scatter_edges(const float* __restrict__ ea,
                                                     const int*   __restrict__ ei,
                                                     float*       __restrict__ es) {
    int e = blockIdx.x;
    int r = ei[NE + e];                         // receiver row of edge_index
    const v4f v = *(const v4f*)(ea + (size_t)e * D + threadIdx.x * 4);
    float* dst = es + (size_t)r * D + threadIdx.x * 4;
    unsafeAtomicAdd(dst + 0, v[0]);
    unsafeAtomicAdd(dst + 1, v[1]);
    unsafeAtomicAdd(dst + 2, v[2]);
    unsafeAtomicAdd(dst + 3, v[3]);
}

// ---------------- kernel 3: weights -> bf16, transposed [out][in] --------
__global__ __launch_bounds__(256) void prep_w(const float* __restrict__ W1,
                                              const float* __restrict__ W2,
                                              __bf16* __restrict__ W1T,
                                              __bf16* __restrict__ W2T) {
    unsigned idx = blockIdx.x * 256 + threadIdx.x;
    BF16CV cv;
    if (idx < (unsigned)(D * IND)) {
        unsigned n = idx >> 10, k = idx & 1023u;       // W1T[n][k] = W1[k][n]
        cv.s = f2bfbits(W1[(size_t)k * D + n]);
        W1T[idx] = cv.b;
    } else if (idx < (unsigned)(D * IND + D * D)) {
        unsigned j = idx - D * IND;
        unsigned n = j >> 9, k = j & 511u;             // W2T[n][k] = W2[k][n]
        cv.s = f2bfbits(W2[(size_t)k * D + n]);
        W2T[j] = cv.b;
    }
}

// ---------------- kernel 4: fused GEMM1+SiLU+GEMM2+LayerNorm -------------
// Block = 256 threads (8 waves), M-tile = 32 nodes.
// Each wave owns 2 M-tiles x 4 N-tiles (64 cols): each B fragment from L2
// feeds two WMMAs, halving weight traffic vs a 16-row tile.
__global__ __launch_bounds__(256) void fused_mlp(const float*  __restrict__ node,
                                                 const float*  __restrict__ esum,
                                                 const __bf16* __restrict__ W1T,
                                                 const __bf16* __restrict__ W2T,
                                                 const float*  __restrict__ b1,
                                                 const float*  __restrict__ b2,
                                                 const float*  __restrict__ gamma,
                                                 const float*  __restrict__ beta,
                                                 float*        __restrict__ out) {
    __shared__ __align__(16) char lds_raw[65536];
    __bf16* xb = (__bf16*)lds_raw;                 // [32][1024] bf16 : 64 KB
    __bf16* hb = (__bf16*)lds_raw;                 // [32][512]  bf16 : overlays dead x
    float*  yb = (float*)lds_raw;                  // [32][512]  f32  : overlays all

    const int tid  = threadIdx.x;
    const int wave = tid >> 5;
    const int lane = tid & 31;
    const int mrow = lane & 15;
    const int hl   = lane >> 4;
    const int m0   = blockIdx.x * 32;

    // ---- stage x = [node || edge_sum] into LDS as bf16 ----
    {
        int row  = tid >> 3;                       // 0..31
        int cseg = (tid & 7) * 128;                // 128 cols per thread
        int gr   = m0 + row; if (gr >= NN) gr = NN - 1;
        const float* nsrc = node + (size_t)gr * D;
        const float* esrc = esum + (size_t)gr * D;
        #pragma unroll
        for (int i = 0; i < 128; i += 4) {
            int c = cseg + i;
            v4f v = (c < D) ? *(const v4f*)(nsrc + c)
                            : *(const v4f*)(esrc + (c - D));
            unsigned p0 = (unsigned)f2bfbits(v[0]) | ((unsigned)f2bfbits(v[1]) << 16);
            unsigned p1 = (unsigned)f2bfbits(v[2]) | ((unsigned)f2bfbits(v[3]) << 16);
            unsigned* dst = (unsigned*)(xb + (size_t)row * IND + c);
            dst[0] = p0; dst[1] = p1;
        }
    }
    __syncthreads();

    v8f zf;
    #pragma unroll
    for (int i = 0; i < 8; ++i) zf[i] = 0.f;

    // ---- GEMM1: (32 x 1024) x (1024 x 512) ----
    v8f acc[2][4] = { { zf, zf, zf, zf }, { zf, zf, zf, zf } };
    {
        const __bf16* w1base = W1T + (size_t)(wave * 64 + mrow) * IND + hl * 8;
        for (int k0 = 0; k0 < IND; k0 += 32) {
            AF a[2];
            #pragma unroll
            for (int g = 0; g < 2; ++g) {
                const __bf16* ap = xb + (size_t)(g * 16 + mrow) * IND + k0 + hl * 8;
                a[g].q[0] = *(const v4u*)ap;
                a[g].q[1] = *(const v4u*)(ap + 16);
            }
            #pragma unroll
            for (int t = 0; t < 4; ++t) {
                AF b;
                const __bf16* bp = w1base + (size_t)t * 16 * IND + k0;
                b.q[0] = *(const v4u*)bp;
                b.q[1] = *(const v4u*)(bp + 16);
                #pragma unroll
                for (int g = 0; g < 2; ++g)
                    acc[g][t] = __builtin_amdgcn_wmma_f32_16x16x32_bf16(
                        false, a[g].v, false, b.v, (short)0, acc[g][t], false, false);
            }
        }
    }
    __syncthreads();                               // all reads of xb done

    // ---- epilogue 1: bias + SiLU -> hb (bf16, overlays first half of x) ----
    #pragma unroll
    for (int t = 0; t < 4; ++t) {
        int n = wave * 64 + t * 16 + mrow;
        float bias = b1[n];
        #pragma unroll
        for (int g = 0; g < 2; ++g) {
            #pragma unroll
            for (int r = 0; r < 8; ++r) {
                int m = g * 16 + hl * 8 + r;
                float v = acc[g][t][r] + bias;
                float s = v / (1.0f + __expf(-v));     // SiLU
                BF16CV cv; cv.s = f2bfbits(s);
                hb[(size_t)m * D + n] = cv.b;
            }
        }
    }
    __syncthreads();

    // ---- GEMM2: (32 x 512) x (512 x 512) ----
    v8f acc2[2][4] = { { zf, zf, zf, zf }, { zf, zf, zf, zf } };
    {
        const __bf16* w2base = W2T + (size_t)(wave * 64 + mrow) * D + hl * 8;
        for (int k0 = 0; k0 < D; k0 += 32) {
            AF a[2];
            #pragma unroll
            for (int g = 0; g < 2; ++g) {
                const __bf16* ap = hb + (size_t)(g * 16 + mrow) * D + k0 + hl * 8;
                a[g].q[0] = *(const v4u*)ap;
                a[g].q[1] = *(const v4u*)(ap + 16);
            }
            #pragma unroll
            for (int t = 0; t < 4; ++t) {
                AF b;
                const __bf16* bp = w2base + (size_t)t * 16 * D + k0;
                b.q[0] = *(const v4u*)bp;
                b.q[1] = *(const v4u*)(bp + 16);
                #pragma unroll
                for (int g = 0; g < 2; ++g)
                    acc2[g][t] = __builtin_amdgcn_wmma_f32_16x16x32_bf16(
                        false, a[g].v, false, b.v, (short)0, acc2[g][t], false, false);
            }
        }
    }
    __syncthreads();                               // all reads of hb done

    // ---- epilogue 2: bias -> yb (f32, overlays x/h region) ----
    #pragma unroll
    for (int t = 0; t < 4; ++t) {
        int n = wave * 64 + t * 16 + mrow;
        float bias = b2[n];
        #pragma unroll
        for (int g = 0; g < 2; ++g) {
            #pragma unroll
            for (int r = 0; r < 8; ++r) {
                int m = g * 16 + hl * 8 + r;
                yb[(size_t)m * D + n] = acc2[g][t][r] + bias;
            }
        }
    }
    __syncthreads();

    // ---- LayerNorm: wave handles 4 rows, lane covers 16 contiguous cols ----
    #pragma unroll
    for (int rr = 0; rr < 4; ++rr) {
        int m = wave * 4 + rr;
        const float* row = yb + (size_t)m * D;
        int cb = lane * 16;
        float vals[16];
        float s = 0.f, s2 = 0.f;
        #pragma unroll
        for (int i = 0; i < 16; ++i) {
            float v = row[cb + i];
            vals[i] = v; s += v; s2 += v * v;
        }
        #pragma unroll
        for (int off = 16; off >= 1; off >>= 1) {
            s  += __shfl_xor(s,  off, 32);
            s2 += __shfl_xor(s2, off, 32);
        }
        float mean = s * (1.f / D);
        float var  = s2 * (1.f / D) - mean * mean;
        float rstd = rsqrtf(var + 1e-5f);
        int gm = m0 + m;
        if (gm < NN) {
            float* orow = out + (size_t)gm * D;
            #pragma unroll
            for (int i = 0; i < 16; ++i) {
                int c = cb + i;
                orow[c] = (vals[i] - mean) * rstd * gamma[c] + beta[c];
            }
        }
    }
}

// -------------------------------------------------------------------------
extern "C" void kernel_launch(void* const* d_in, const int* in_sizes, int n_in,
                              void* d_out, int out_size, void* d_ws, size_t ws_size,
                              hipStream_t stream) {
    const float* node  = (const float*)d_in[0];
    const int*   ei    = (const int*)d_in[1];
    const float* ea    = (const float*)d_in[2];
    const float* W1    = (const float*)d_in[3];
    const float* b1    = (const float*)d_in[4];
    const float* W2    = (const float*)d_in[5];
    const float* b2    = (const float*)d_in[6];
    const float* gamma = (const float*)d_in[7];
    const float* beta  = (const float*)d_in[8];
    float* out = (float*)d_out;

    const size_t w1sz = (size_t)D * IND * sizeof(unsigned short);
    const size_t w2sz = (size_t)D * D   * sizeof(unsigned short);
    const size_t essz = (size_t)NN * D  * sizeof(float);

    char* ws = (char*)d_ws;
    __bf16* W1T = (__bf16*)ws;
    __bf16* W2T = (__bf16*)(ws + w1sz);
    // edge_sum accumulator: prefer workspace, else safely reuse d_out
    float* es = (ws_size >= w1sz + w2sz + essz) ? (float*)(ws + w1sz + w2sz)
                                                : out;

    const int zeroBlocks = (int)(((size_t)NN * D / 4 + 255) / 256);
    zero_f32<<<zeroBlocks, 256, 0, stream>>>(es);

    const int prepBlocks = (D * IND + D * D + 255) / 256;
    prep_w<<<prepBlocks, 256, 0, stream>>>(W1, W2, W1T, W2T);

    scatter_edges<<<NE, 128, 0, stream>>>(ea, ei, es);

    fused_mlp<<<(NN + 31) / 32, 256, 0, stream>>>(node, es, W1T, W2T,
                                                  b1, b2, gamma, beta, out);
}
